// CapsuleNet_33509334843687
// MI455X (gfx1250) — matrix operand order repaired and checked by reading, not compiled, not run on hardware
//
#include <hip/hip_runtime.h>
#include <hip/hip_bf16.h>
#include <math.h>

// ---------------------------------------------------------------------------
// CapsuleNet forward for MI455X (gfx1250, wave32, WMMA).
//  conv1 / prim-conv : implicit GEMM, v_wmma_f32_16x16x32_bf16, f32 accum
//                      (weights zero-padded so A loads are unconditional
//                       16B vector loads; im2col decode hoisted out of K loop)
//  routing           : W (472 MB) streamed from HBM each pass; u_hat
//                      recomputed inside WMMA GEMMs (HBM-bound)
//  decoder MLPs      : tiny (M=8), plain VALU dot products
// ---------------------------------------------------------------------------

typedef __attribute__((ext_vector_type(16))) __bf16 v16bf;
typedef __attribute__((ext_vector_type(8)))  __bf16 v8bf;
typedef __attribute__((ext_vector_type(8)))  float  v8f;

#define NCAPS 18432   // in_caps
#define NCLS  50

// ---------------- small utility kernels ----------------
__global__ void k_cvt_bf16(const float* __restrict__ s, __bf16* __restrict__ d, int n) {
  int i = blockIdx.x * blockDim.x + threadIdx.x;
  if (i < n) d[i] = (__bf16)s[i];
}

// convert [rows][K] f32 -> [rows][Kpad] bf16, zero-filling k >= K
__global__ void k_cvt_pad_bf16(const float* __restrict__ s, __bf16* __restrict__ d,
                               int rows, int K, int Kpad) {
  int i = blockIdx.x * blockDim.x + threadIdx.x;
  if (i >= rows * Kpad) return;
  const int r = i / Kpad, k = i % Kpad;
  d[i] = (k < K) ? (__bf16)s[(size_t)r * K + k] : (__bf16)0.0f;
}

__global__ void k_zero_f32(float* __restrict__ p, int n) {
  int i = blockIdx.x * blockDim.x + threadIdx.x;
  if (i < n) p[i] = 0.f;
}

// ---------------- conv1: 3->256, 9x9, valid, stride 1 ----------------
// GEMM view: M=256 (co), K=3*81=243 (zero-padded to 256), N=8*56*56=25088.
// One wave per 16x16 output tile.
__global__ __launch_bounds__(32) void k_conv1(
    const __bf16* __restrict__ xb, const __bf16* __restrict__ w1b,
    const float* __restrict__ bias, __bf16* __restrict__ hb)
{
  const int l = threadIdx.x, lh = l & 15, hi = l >> 4;
  const int m0 = blockIdx.x * 16;
  const int n0 = blockIdx.y * 16;
  v8f acc = {};
  const __bf16* wrow = w1b + (size_t)(m0 + lh) * 256;   // padded K stride
  const __bf16 z = (__bf16)0.0f;

  // hoist im2col column bases (wave-uniform, K-invariant):
  // addr(k,j) = obase[j] + c*4096 + ky*64 + kx
  int obase[16];
#pragma unroll
  for (int j = 0; j < 16; ++j) {
    const int n = n0 + j;
    const int b = n / 3136, rem = n % 3136, y = rem / 56, x = rem % 56;
    obase[j] = ((b * 3) * 64 + y) * 64 + x;
  }

  for (int k0 = 0; k0 < 256; k0 += 32) {
    // A: two unconditional 16B loads (zero-padded weights)
    v16bf a;
    {
      const v8bf alo = *(const v8bf*)(wrow + k0 + (hi ? 8 : 0));
      const v8bf ahi = *(const v8bf*)(wrow + k0 + (hi ? 24 : 16));
#pragma unroll
      for (int j = 0; j < 8; ++j) { a[j] = alo[j]; a[8 + j] = ahi[j]; }
    }
    // B: lane = column n0+lh; element j -> k in table order
    v16bf bm;
#pragma unroll
    for (int j = 0; j < 16; ++j) {
      const int kg = k0 + (j < 8 ? 8 * hi + j : 16 + 8 * hi + (j - 8));
      if (kg < 243) {
        const int c = kg / 81, rr = kg % 81, ky = rr / 9, kx = rr % 9;
        const int koff = c * 4096 + ky * 64 + kx;
        bm[j] = xb[obase[lh] + koff];
      } else {
        bm[j] = z;
      }
    }
    acc = __builtin_amdgcn_wmma_f32_16x16x32_bf16(false, a, false, bm, (short)0, acc, false, false);
  }
  // D: lane column n = n0+lh, rows m = r + 8*hi
  const int n = n0 + lh;
  const int b = n / 3136, rem = n % 3136, y = rem / 56, x = rem % 56;
#pragma unroll
  for (int r = 0; r < 8; ++r) {
    const int co = m0 + r + (hi ? 8 : 0);
    const float f = fmaxf(acc[r] + bias[co], 0.f);   // + bias, ReLU
    hb[((size_t)(b * 256 + co) * 56 + y) * 56 + x] = (__bf16)f;
  }
}

// ---------------- prim conv: 256->256, 9x9, stride 2 ----------------
// GEMM view: M=256, K=256*81=20736 (exact multiple of 32), N=8*24*24=4608.
__global__ __launch_bounds__(32) void k_conv2(
    const __bf16* __restrict__ hb, const __bf16* __restrict__ w2b,
    const float* __restrict__ bias, float* __restrict__ p)
{
  const int l = threadIdx.x, lh = l & 15, hi = l >> 4;
  const int m0 = blockIdx.x * 16;
  const int n0 = blockIdx.y * 16;
  v8f acc = {};
  const __bf16* wrow = w2b + (size_t)(m0 + lh) * 20736;

  // hoist im2col column bases: addr(k,j) = obase[j] + c*3136 + ky*56 + kx
  int obase[16];
#pragma unroll
  for (int j = 0; j < 16; ++j) {
    const int n = n0 + j;
    const int b = n / 576, rem = n % 576, y = rem / 24, x = rem % 24;
    obase[j] = ((b * 256) * 56 + 2 * y) * 56 + 2 * x;
  }

  for (int k0 = 0; k0 < 20736; k0 += 32) {
    v16bf a;
    {
      const v8bf alo = *(const v8bf*)(wrow + k0 + (hi ? 8 : 0));
      const v8bf ahi = *(const v8bf*)(wrow + k0 + (hi ? 24 : 16));
#pragma unroll
      for (int j = 0; j < 8; ++j) { a[j] = alo[j]; a[8 + j] = ahi[j]; }
    }
    v16bf bm;
#pragma unroll
    for (int j = 0; j < 16; ++j) {
      const int kg = k0 + (j < 8 ? 8 * hi + j : 16 + 8 * hi + (j - 8));
      const int c = kg / 81, rr = kg % 81, ky = rr / 9, kx = rr % 9;
      const int koff = c * 3136 + ky * 56 + kx;
      bm[j] = hb[obase[lh] + koff];
    }
    acc = __builtin_amdgcn_wmma_f32_16x16x32_bf16(false, a, false, bm, (short)0, acc, false, false);
  }
  const int n = n0 + lh;
  const int b = n / 576, rem = n % 576, y = rem / 24, x = rem % 24;
#pragma unroll
  for (int r = 0; r < 8; ++r) {
    const int co = m0 + r + (hi ? 8 : 0);
    p[((size_t)(b * 256 + co) * 24 + y) * 24 + x] = acc[r] + bias[co];
  }
}

// ---------------- primary capsule squash: p[8,256,24,24] -> u[8,18432,8] ----
__global__ void k_squash_u(const float* __restrict__ p, float* __restrict__ u) {
  const int t = blockIdx.x * blockDim.x + threadIdx.x;
  if (t >= 8 * NCAPS) return;
  const int b = t / NCAPS, i = t % NCAPS;
  const int cg = i / 576, sp = i % 576;
  float e[8], sn = 0.f;
#pragma unroll
  for (int j = 0; j < 8; ++j) {
    e[j] = p[(size_t)(b * 256 + cg * 8 + j) * 576 + sp];
    sn += e[j] * e[j];
  }
  const float s = sn / ((1.f + sn) * (sqrtf(sn) + 1e-8f));
#pragma unroll
  for (int j = 0; j < 8; ++j) u[(size_t)t * 8 + j] = e[j] * s;
}

// ---------------- routing pass A: v_num[b,n,d] = sum_{i,e} W*(sc*u) ---------
// WMMA GEMM: M=16 (d), N=16 (8 batches + pad), K=(i,e) chunked 32 (4 caps).
// Streams W (472 MB) from HBM; deterministic partials per (n, slab).
__global__ __launch_bounds__(32) void k_route_vnum(
    const float* __restrict__ W, const float* __restrict__ u,
    const float* __restrict__ blog, const float* __restrict__ mx,
    const float* __restrict__ invden, float* __restrict__ part, int uniform)
{
  const int l = threadIdx.x, lh = l & 15, hi = l >> 4;
  const int n = blockIdx.y;
  const int chunks = 4608 / gridDim.x;
  const int c0 = blockIdx.x * chunks;
  const int iiA = hi ? 1 : 0, iiB = hi ? 3 : 2;   // A-operand capsule sub-index
  const int eB = l & 7, iiBl = l >> 3;            // B-operand row (k = l)
  const __bf16 z = (__bf16)0.0f;
  v8f acc = {};

  for (int cc = 0; cc < chunks; ++cc) {
    const int i0 = (c0 + cc) * 4;
    // A: W[n, i0+ii, d=lh, e=0..7] -- two coalesced 32B runs per lane
    const float* pa = W + (((size_t)n * NCAPS + i0 + iiA) * 16 + lh) * 8;
    const float* pb = W + (((size_t)n * NCAPS + i0 + iiB) * 16 + lh) * 8;
    __builtin_prefetch(pa + 512, 0, 1);           // next chunk of the W stream
    const v8f wa = *(const v8f*)pa;
    const v8f wb = *(const v8f*)pb;
    v16bf a;
#pragma unroll
    for (int j = 0; j < 8; ++j) { a[j] = (__bf16)wa[j]; a[8 + j] = (__bf16)wb[j]; }
    // B: col b = lh (batch, valid < 8); element s -> (ii,e) in table order
    v16bf bm;
#pragma unroll
    for (int s = 0; s < 16; ++s) {
      const int ib = i0 + (s < 8 ? iiA : iiB), es = s & 7, b = lh;
      if (b < 8) {
        float sc;
        if (uniform) sc = 0.02f;  // softmax of zeros over 50 classes
        else {
          const float bl = blog[((size_t)b * NCLS + n) * NCAPS + ib];
          sc = __expf(bl - mx[b * NCAPS + ib]) * invden[b * NCAPS + ib];
        }
        bm[s] = (__bf16)(sc * u[((size_t)b * NCAPS + ib) * 8 + es]);
      } else {
        bm[s] = z;
      }
    }
    acc = __builtin_amdgcn_wmma_f32_16x16x32_bf16(false, a, false, bm, (short)0, acc, false, false);
  }
  // D: lane col b = lh (valid < 8), rows d = r + 8*hi; write partial [n][slab][b][d]
  if (lh < 8) {
#pragma unroll
    for (int r = 0; r < 8; ++r)
      part[(((size_t)n * gridDim.x + blockIdx.x) * 8 + lh) * 16 + r + (hi ? 8 : 0)] = acc[r];
  }
}

__global__ void k_vnum_reduce(const float* __restrict__ part, float* __restrict__ vnum, int nslabs) {
  const int t = blockIdx.x * blockDim.x + threadIdx.x;
  if (t >= 8 * NCLS * 16) return;
  const int b = t / (NCLS * 16), rem = t % (NCLS * 16), n = rem / 16, d = rem % 16;
  float s = 0.f;
  for (int sl = 0; sl < nslabs; ++sl)
    s += part[(((size_t)n * nslabs + sl) * 8 + b) * 16 + d];
  vnum[t] = s;   // layout [b][n][d]
}

__global__ void k_squash_v(const float* __restrict__ vn, float* __restrict__ v) {
  const int t = blockIdx.x * blockDim.x + threadIdx.x;
  if (t >= 8 * NCLS) return;
  float e[16], sn = 0.f;
#pragma unroll
  for (int d = 0; d < 16; ++d) { e[d] = vn[t * 16 + d]; sn += e[d] * e[d]; }
  const float s = sn / ((1.f + sn) * (sqrtf(sn) + 1e-8f));
#pragma unroll
  for (int d = 0; d < 16; ++d) v[t * 16 + d] = e[d] * s;
}

// ---------------- routing pass B: b_log[b,n,i] += u_hat . v -----------------
// T[(i2,e),b] = sum_d W[n,i,d,e]*v[b,n,d] via WMMA (K=16 zero-padded to 32),
// then epilogue upd = sum_e u[b,i,e]*T. One wave per (n, i-pair stream).
__global__ __launch_bounds__(32) void k_route_blog(
    const float* __restrict__ W, const float* __restrict__ u,
    const float* __restrict__ v, float* __restrict__ blog)
{
  const int l = threadIdx.x, lh = l & 15, hi = l >> 4;
  const int n = blockIdx.y;
  const int chunks = 9216 / gridDim.x;
  const int c0 = blockIdx.x * chunks;
  const __bf16 z = (__bf16)0.0f;

  // B operand is loop-invariant per n: col b = lh (<8), element s<8 -> k=d=8*hi+s, s>=8 zero
  v16bf bm;
#pragma unroll
  for (int s = 0; s < 16; ++s) {
    float val = 0.f;
    if (lh < 8 && s < 8) val = v[((size_t)lh * NCLS + n) * 16 + (hi ? 8 : 0) + s];
    bm[s] = (__bf16)val;
  }
  const int i2 = lh >> 3, e = lh & 7;   // A-operand row index (m = i2*8+e)

  for (int cc = 0; cc < chunks; ++cc) {
    const int i0 = (c0 + cc) * 2;
    const float* wbase = W + (((size_t)n * NCAPS + i0 + i2) * 16) * 8 + e;
    __builtin_prefetch(wbase + 256, 0, 1);
    v16bf a;
#pragma unroll
    for (int j = 0; j < 8; ++j) {
      const int d = (hi ? 8 : 0) + j;
      a[j] = (__bf16)wbase[(size_t)d * 8];  // k = 0..15 -> d
      a[8 + j] = z;                         // k = 16..31 zero pad
    }
    v8f acc = {};
    acc = __builtin_amdgcn_wmma_f32_16x16x32_bf16(false, a, false, bm, (short)0, acc, false, false);
    // D rows m = r + 8*hi = (i2=hi, e=r); col b = lh
    if (lh < 8) {
      const int i = i0 + hi;
      const float* ur = u + ((size_t)lh * NCAPS + i) * 8;
      float upd = 0.f;
#pragma unroll
      for (int r = 0; r < 8; ++r) upd = fmaf(ur[r], acc[r], upd);
      blog[((size_t)lh * NCLS + n) * NCAPS + i] += upd;   // unique lane -> no atomics
    }
  }
}

// ---------------- softmax stats over classes (per (b,i)) --------------------
__global__ void k_stats(const float* __restrict__ blog, float* __restrict__ mx,
                        float* __restrict__ invden) {
  const int t = blockIdx.x * blockDim.x + threadIdx.x;
  if (t >= 8 * NCAPS) return;
  const int b = t / NCAPS, i = t % NCAPS;
  const float* base = blog + (size_t)b * NCLS * NCAPS + i;
  float m = -1e30f;
  for (int n = 0; n < NCLS; ++n) m = fmaxf(m, base[(size_t)n * NCAPS]);
  float s = 0.f;
  for (int n = 0; n < NCLS; ++n) s += __expf(base[(size_t)n * NCAPS] - m);
  mx[t] = m;
  invden[t] = 1.f / s;
}

// ---------------- predictions + argmax mask ---------------------------------
__global__ void k_preds(const float* __restrict__ v, float* __restrict__ out) {
  const int t = blockIdx.x * blockDim.x + threadIdx.x;
  if (t >= 8 * NCLS) return;
  float sn = 0.f;
#pragma unroll
  for (int d = 0; d < 16; ++d) { const float x = v[t * 16 + d]; sn += x * x; }
  out[t] = sqrtf(sn);
}

__global__ __launch_bounds__(64) void k_mask(const float* __restrict__ preds,
                                             const float* __restrict__ v,
                                             float* __restrict__ r0) {
  const int b = blockIdx.x, t = threadIdx.x;
  __shared__ float sv[64];
  __shared__ int   si[64];
  float best = -1e30f; int bi = 0x7fffffff;
  for (int n = t; n < NCLS; n += 64) {
    const float pv = preds[b * NCLS + n];
    if (pv > best) { best = pv; bi = n; }
  }
  sv[t] = best; si[t] = bi;
  __syncthreads();
  for (int s = 32; s > 0; s >>= 1) {
    if (t < s) {
      if (sv[t + s] > sv[t] || (sv[t + s] == sv[t] && si[t + s] < si[t])) {
        sv[t] = sv[t + s]; si[t] = si[t + s];
      }
    }
    __syncthreads();
  }
  const int nb = si[0];
  for (int j = t; j < 800; j += 64) {
    const int n = j >> 4, d = j & 15;
    r0[b * 800 + j] = (n == nb) ? v[(b * NCLS + n) * 16 + d] : 0.f;
  }
}

// ---------------- decoder fully-connected layers ----------------------------
__global__ void k_fc(const float* __restrict__ in, const float* __restrict__ w,
                     const float* __restrict__ bias, float* __restrict__ out,
                     int B, int K, int O, int act) {
  const int t = blockIdx.x * blockDim.x + threadIdx.x;
  if (t >= B * O) return;
  const int b = t / O, o = t % O;
  const float* wr = w + (size_t)o * K;
  const float* ir = in + (size_t)b * K;
  float s = bias[o];
  for (int k = 0; k < K; ++k) s = fmaf(ir[k], wr[k], s);
  out[(size_t)b * O + o] = (act == 0) ? fmaxf(s, 0.f) : 1.f / (1.f + __expf(-s));
}

// ---------------------------------------------------------------------------
extern "C" void kernel_launch(void* const* d_in, const int* in_sizes, int n_in,
                              void* d_out, int out_size, void* d_ws, size_t ws_size,
                              hipStream_t stream) {
  (void)in_sizes; (void)n_in; (void)out_size; (void)ws_size;
  const float* x   = (const float*)d_in[0];
  const float* w1  = (const float*)d_in[1];
  const float* b1  = (const float*)d_in[2];
  const float* w2  = (const float*)d_in[3];
  const float* b2  = (const float*)d_in[4];
  const float* W   = (const float*)d_in[5];
  const float* d1w = (const float*)d_in[6];
  const float* d1b = (const float*)d_in[7];
  const float* d2w = (const float*)d_in[8];
  const float* d2b = (const float*)d_in[9];
  const float* d3w = (const float*)d_in[10];
  const float* d3b = (const float*)d_in[11];
  float* out = (float*)d_out;

  char* ws = (char*)d_ws;
  size_t off = 0;
  auto alloc = [&](size_t bytes) -> void* {
    void* p = ws + off;
    off = (off + bytes + 255) & ~(size_t)255;
    return p;
  };
  __bf16* xb   = (__bf16*)alloc((size_t)98304 * 2);
  __bf16* w1b  = (__bf16*)alloc((size_t)256 * 256 * 2);   // K padded 243->256
  __bf16* w2b  = (__bf16*)alloc((size_t)5308416 * 2);
  __bf16* hb   = (__bf16*)alloc((size_t)6422528 * 2);     // conv1 out (bf16)
  float*  p    = (float*)alloc((size_t)1179648 * 4);      // prim conv out
  float*  u    = (float*)alloc((size_t)1179648 * 4);      // squashed capsules
  float*  blog = (float*)alloc((size_t)7372800 * 4);      // routing logits
  float*  mx   = (float*)alloc((size_t)147456 * 4);
  float*  invd = (float*)alloc((size_t)147456 * 4);
  float*  part = (float*)alloc((size_t)NCLS * 48 * 128 * 4);
  float*  vnum = (float*)alloc((size_t)6400 * 4);
  float*  v    = (float*)alloc((size_t)6400 * 4);
  float*  r0   = (float*)alloc((size_t)6400 * 4);
  float*  r1   = (float*)alloc((size_t)4096 * 4);
  float*  r2   = (float*)alloc((size_t)8192 * 4);
  // total ws use ~= 64.5 MB

  auto cdiv = [](int a, int b) { return (a + b - 1) / b; };
  const int NSLAB_V = 48, NSLAB_B = 96;

  // precision staging (bf16 operands for WMMA)
  k_cvt_bf16<<<cdiv(98304, 256), 256, 0, stream>>>(x, xb, 98304);
  k_cvt_pad_bf16<<<cdiv(256 * 256, 256), 256, 0, stream>>>(w1, w1b, 256, 243, 256);
  k_cvt_bf16<<<cdiv(5308416, 256), 256, 0, stream>>>(w2, w2b, 5308416);

  // convolutions (WMMA implicit GEMM) + primary-capsule squash
  k_conv1<<<dim3(16, 1568), 32, 0, stream>>>(xb, w1b, b1, hb);
  k_conv2<<<dim3(16, 288), 32, 0, stream>>>(hb, w2b, b2, p);
  k_squash_u<<<cdiv(8 * NCAPS, 256), 256, 0, stream>>>(p, u);

  k_zero_f32<<<cdiv(7372800, 256), 256, 0, stream>>>(blog, 7372800);

  // routing iteration 1 (uniform coupling: softmax of zero logits)
  k_route_vnum<<<dim3(NSLAB_V, NCLS), 32, 0, stream>>>(W, u, blog, mx, invd, part, 1);
  k_vnum_reduce<<<cdiv(6400, 256), 256, 0, stream>>>(part, vnum, NSLAB_V);
  k_squash_v<<<cdiv(400, 256), 256, 0, stream>>>(vnum, v);
  k_route_blog<<<dim3(NSLAB_B, NCLS), 32, 0, stream>>>(W, u, v, blog);

  // routing iteration 2
  k_stats<<<cdiv(8 * NCAPS, 256), 256, 0, stream>>>(blog, mx, invd);
  k_route_vnum<<<dim3(NSLAB_V, NCLS), 32, 0, stream>>>(W, u, blog, mx, invd, part, 0);
  k_vnum_reduce<<<cdiv(6400, 256), 256, 0, stream>>>(part, vnum, NSLAB_V);
  k_squash_v<<<cdiv(400, 256), 256, 0, stream>>>(vnum, v);
  k_route_blog<<<dim3(NSLAB_B, NCLS), 32, 0, stream>>>(W, u, v, blog);

  // final agreement
  k_stats<<<cdiv(8 * NCAPS, 256), 256, 0, stream>>>(blog, mx, invd);
  k_route_vnum<<<dim3(NSLAB_V, NCLS), 32, 0, stream>>>(W, u, blog, mx, invd, part, 0);
  k_vnum_reduce<<<cdiv(6400, 256), 256, 0, stream>>>(part, vnum, NSLAB_V);
  k_squash_v<<<cdiv(400, 256), 256, 0, stream>>>(vnum, v);

  // outputs: preds -> out[0..400), recon -> out[400..)
  k_preds<<<cdiv(400, 256), 256, 0, stream>>>(v, out);
  k_mask<<<8, 64, 0, stream>>>(out, v, r0);
  k_fc<<<cdiv(8 * 512, 256), 256, 0, stream>>>(r0, d1w, d1b, r1, 8, 800, 512, 0);
  k_fc<<<cdiv(8 * 1024, 256), 256, 0, stream>>>(r1, d2w, d2b, r2, 8, 512, 1024, 0);
  k_fc<<<cdiv(8 * 12288, 256), 256, 0, stream>>>(r2, d3w, d3b, out + 400, 8, 1024, 12288, 1);
}
